// q_MixerBlock_30562987278757
// MI455X (gfx1250) — 8-bit matrix operand order repaired and checked by reading, compiled, not run on hardware
//
#include <hip/hip_runtime.h>
#include <math.h>


typedef __attribute__((ext_vector_type(8))) int v8i;

#define DEV __device__ __forceinline__

#ifndef __has_builtin
#define __has_builtin(x) 0
#endif
#if __has_builtin(__builtin_amdgcn_global_load_async_to_lds_b128) && \
    __has_builtin(__builtin_amdgcn_s_wait_asynccnt)
#define USE_ASYNC_COPY 1
#else
#define USE_ASYNC_COPY 0
#endif

#if USE_ASYNC_COPY
typedef int v4i_raw __attribute__((vector_size(16)));
typedef v4i_raw __attribute__((address_space(1)))* as1_v4p;
typedef v4i_raw __attribute__((address_space(3)))* as3_v4p;
#endif

constexpr int Bb  = 128;
constexpr int Ss  = 196;
constexpr int Dd  = 768;
constexpr int SP  = 256;
constexpr int BS  = Bb * Ss;
constexpr int BD  = Bb * Dd;
constexpr long long NTOT = (long long)Bb * Ss * Dd;

DEV float sf_from_abs(float a) { return fmaxf(a, 1e-8f) * (1.0f / 127.0f); }
DEV float clip8f(float q)      { return fminf(127.0f, fmaxf(-128.0f, q)); }
DEV void  atomicMaxPosF(float* p, float v) { atomicMax((int*)p, __float_as_int(v)); }
DEV float gelu_exact(float x)  { return 0.5f * x * (1.0f + erff(x * 0.70710678118654752f)); }

DEV float blockMax256(float v, float* red) {
  int tid = threadIdx.x;
  red[tid] = v;
  __syncthreads();
  #pragma unroll
  for (int st = 128; st > 0; st >>= 1) {
    if (tid < st) red[tid] = fmaxf(red[tid], red[tid + st]);
    __syncthreads();
  }
  float r = red[0];
  __syncthreads();
  return r;
}

DEV int a_frag_k(int l, int j, int b) {
  return (j >> 2) * 32 + (l >> 4) * 16 + (j & 3) * 4 + b;
}

__global__ void k_zero_scal(float* scal) {
  if (threadIdx.x < 16) scal[threadIdx.x] = 0.0f;
}

__global__ __launch_bounds__(256) void k_ln_absmax(
    const float* __restrict__ x, const float* __restrict__ g,
    const float* __restrict__ b, float* __restrict__ out,
    float* __restrict__ slot)
{
  __shared__ float red[256];
  const int row = blockIdx.x;
  const int tid = threadIdx.x;
  const float* xr = x + (long long)row * Dd;

  float v0 = xr[tid], v1 = xr[tid + 256], v2 = xr[tid + 512];
  float s = v0 + v1 + v2;
  red[tid] = s; __syncthreads();
  #pragma unroll
  for (int st = 128; st > 0; st >>= 1) { if (tid < st) red[tid] += red[tid + st]; __syncthreads(); }
  float mu = red[0] * (1.0f / (float)Dd);
  __syncthreads();

  float d0 = v0 - mu, d1 = v1 - mu, d2 = v2 - mu;
  red[tid] = d0*d0 + d1*d1 + d2*d2; __syncthreads();
  #pragma unroll
  for (int st = 128; st > 0; st >>= 1) { if (tid < st) red[tid] += red[tid + st]; __syncthreads(); }
  float rstd = rsqrtf(red[0] * (1.0f / (float)Dd) + 1e-6f);
  __syncthreads();

  float y0 = d0 * rstd * g[tid]       + b[tid];
  float y1 = d1 * rstd * g[tid + 256] + b[tid + 256];
  float y2 = d2 * rstd * g[tid + 512] + b[tid + 512];

  float* orow = out + (long long)row * Dd;
  orow[tid] = y0; orow[tid + 256] = y1; orow[tid + 512] = y2;

  float am = fmaxf(fabsf(y0), fmaxf(fabsf(y1), fabsf(y2)));
  float bm = blockMax256(am, red);
  if (tid == 0) atomicMaxPosF(slot, bm);
}

__global__ __launch_bounds__(256) void k_quantA_token(
    const float* __restrict__ ln, const float* __restrict__ absIn,
    signed char* __restrict__ qa)
{
  long long idx = (long long)blockIdx.x * 256 + threadIdx.x;
  const int KT = SP >> 6;
  long long block = idx >> 10;
  int within = (int)(idx & 1023);
  int l = within >> 5, r = within & 31, j = r >> 2, by = r & 3;
  long long mt = block / KT;
  int kt = (int)(block % KT);
  long long m = mt * 16 + (l & 15);
  int k = kt * 64 + a_frag_k(l, j, by);
  int b = (int)(m / Dd);
  int d = (int)(m % Dd);
  float sf = sf_from_abs(absIn[0]);
  float v = 0.0f;
  if (k < Ss) v = ln[((long long)b * Ss + k) * Dd + d];
  qa[idx] = (signed char)clip8f(rintf(v / sf));
}

__global__ __launch_bounds__(256) void k_quantA_flat(
    const float* __restrict__ src, const float* __restrict__ absIn,
    signed char* __restrict__ qa, int KT, long long nbytes)
{
  long long idx = (long long)blockIdx.x * 256 + threadIdx.x;
  if (idx >= nbytes) return;
  long long block = idx >> 10;
  int within = (int)(idx & 1023);
  int l = within >> 5, r = within & 31, j = r >> 2, by = r & 3;
  long long mt = block / KT;
  int kt = (int)(block % KT);
  long long m = mt * 16 + (l & 15);
  int k = kt * 64 + a_frag_k(l, j, by);
  int Kp = KT << 6;
  float sf = sf_from_abs(absIn[0]);
  float v = src[m * Kp + k];
  qa[idx] = (signed char)clip8f(rintf(v / sf));
}

__global__ __launch_bounds__(256) void k_wquant(
    const float* __restrict__ W, int Kin, int Kp, int Np, int Nout,
    signed char* __restrict__ qw, float* __restrict__ wsf)
{
  __shared__ float red[256];
  const int n = blockIdx.x;
  const int tid = threadIdx.x;
  const float* wr = W + (long long)n * Kin;

  float m = 0.0f;
  if (n < Nout)
    for (int k = tid; k < Kin; k += 256) m = fmaxf(m, fabsf(wr[k]));
  float bm = blockMax256(m, red);

  float s   = (n < Nout) ? fmaxf(bm, 1e-8f) * (1.0f / 127.0f) : 0.0f;
  float inv = (n < Nout) ? 127.0f / fmaxf(bm, 1e-8f) : 0.0f;
  if (tid == 0) wsf[n] = s;

  const int NT = Np >> 4;
  for (int k = tid; k < Kp; k += 256) {
    float q = 0.0f;
    if (n < Nout && k < Kin) q = clip8f(rintf(wr[k] * inv));
    int kt   = k >> 6;
    int kk   = k & 63;
    int half = kk >> 5;
    int rem  = kk & 31;
    int kh   = rem >> 4;
    int r2   = rem & 15;
    int j    = half * 4 + (r2 >> 2);
    int by   = r2 & 3;
    long long addr = ((long long)kt * NT + (n >> 4)) * 1024
                   + (kh * 16 + (n & 15)) * 32 + j * 4 + by;
    qw[addr] = (signed char)q;
  }
}

__global__ __launch_bounds__(256) void k_gemm_i8(
    const signed char* __restrict__ Afrag, const signed char* __restrict__ Bfrag,
    const float* __restrict__ bias, const float* __restrict__ wsf,
    const float* __restrict__ preAbs, float* __restrict__ Out,
    float* __restrict__ absSlot, int Kp, int Np)
{
  __shared__ int lA[2048];
  __shared__ int lB[2048];

  const int tid  = threadIdx.x;
  const int lane = tid & 31;
  const int wid  = tid >> 5;
  const int wm   = wid & 3;
  const int wn   = wid >> 2;
  const long long bm = (long long)blockIdx.y * 128;
  const int bn = blockIdx.x * 128;
  const int KT = Kp >> 6;
  const int NT = Np >> 4;

  const v8i zero8 = {0, 0, 0, 0, 0, 0, 0, 0};
  v8i acc[2][4];
  #pragma unroll
  for (int i = 0; i < 2; ++i)
    #pragma unroll
    for (int j = 0; j < 4; ++j) acc[i][j] = zero8;

  const signed char* gA = Afrag + (((long long)blockIdx.y * 8 + wid) * KT) * 1024 + lane * 32;
  const signed char* gB = Bfrag + ((long long)(blockIdx.x * 8 + wid)) * 1024 + lane * 32;
  int* la = &lA[wid * 256 + lane * 8];
  int* lb = &lB[wid * 256 + lane * 8];

  for (int kt = 0; kt < KT; ++kt) {
    const signed char* ga = gA + (long long)kt * 1024;
    const signed char* gb = gB + (long long)kt * NT * 1024;
    __syncthreads();
#if USE_ASYNC_COPY
    __builtin_amdgcn_global_load_async_to_lds_b128(
        (as1_v4p)(signed char*)ga, (as3_v4p)la, 0, 0);
    __builtin_amdgcn_global_load_async_to_lds_b128(
        (as1_v4p)(signed char*)ga, (as3_v4p)la, 16, 0);
    __builtin_amdgcn_global_load_async_to_lds_b128(
        (as1_v4p)(signed char*)gb, (as3_v4p)lb, 0, 0);
    __builtin_amdgcn_global_load_async_to_lds_b128(
        (as1_v4p)(signed char*)gb, (as3_v4p)lb, 16, 0);
    __builtin_amdgcn_s_wait_asynccnt(0);
#else
    {
      const int4 a0 = *(const int4*)ga;
      const int4 a1 = *(const int4*)(ga + 16);
      const int4 b0 = *(const int4*)gb;
      const int4 b1 = *(const int4*)(gb + 16);
      *(int4*)la       = a0;
      *(int4*)(la + 4) = a1;
      *(int4*)lb       = b0;
      *(int4*)(lb + 4) = b1;
    }
#endif
    if (kt + 1 < KT) {
      __builtin_prefetch(ga + 1024, 0, 0);
      __builtin_prefetch(gb + (long long)NT * 1024, 0, 0);
    }
    __syncthreads();

    v8i a0 = *(const v8i*)&lA[(wm * 2 + 0) * 256 + lane * 8];
    v8i a1 = *(const v8i*)&lA[(wm * 2 + 1) * 256 + lane * 8];
    #pragma unroll
    for (int jn = 0; jn < 4; ++jn) {
      v8i bf = *(const v8i*)&lB[(wn * 4 + jn) * 256 + lane * 8];
      acc[0][jn] = __builtin_amdgcn_wmma_i32_16x16x64_iu8(true, a0, true, bf,
                                                          acc[0][jn], false, false);
      acc[1][jn] = __builtin_amdgcn_wmma_i32_16x16x64_iu8(true, a1, true, bf,
                                                          acc[1][jn], false, false);
    }
  }

  const float pre = sf_from_abs(preAbs[0]);
  const int nloc = lane & 15;
  const int mhi  = (lane >> 4) * 8;
  float am = 0.0f;
  #pragma unroll
  for (int i = 0; i < 2; ++i) {
    #pragma unroll
    for (int jn = 0; jn < 4; ++jn) {
      int n = bn + wn * 64 + jn * 16 + nloc;
      float ws  = wsf[n];
      float bsf = pre * ws;
      float bq  = (ws > 0.0f) ? rintf(bias[(ws > 0.0f) ? n : 0] / bsf) : 0.0f;
      #pragma unroll
      for (int r = 0; r < 8; ++r) {
        long long m = bm + wm * 32 + i * 16 + mhi + r;
        float o = (ws > 0.0f) ? ((float)acc[i][jn][r] + bq) * bsf : 0.0f;
        Out[m * Np + n] = o;
        am = fmaxf(am, fabsf(o));
      }
    }
  }
  atomicMaxPosF(absSlot, am);
}

__global__ __launch_bounds__(256) void k_requant_gelu(
    float* __restrict__ h, const float* __restrict__ absIn,
    float* __restrict__ absOut, long long n)
{
  __shared__ float red[256];
  long long idx = (long long)blockIdx.x * 256 + threadIdx.x;
  float sf = sf_from_abs(absIn[0]);
  float am = 0.0f;
  if (idx < n) {
    float q = clip8f(rintf(h[idx] / sf)) * sf;
    float g = gelu_exact(q);
    h[idx] = g;
    am = fabsf(g);
  }
  float bm = blockMax256(am, red);
  if (threadIdx.x == 0) atomicMaxPosF(absOut, bm);
}

__global__ __launch_bounds__(256) void k_resid_absmax(
    const float* __restrict__ h, const float* __restrict__ ident,
    float* __restrict__ slot, int token)
{
  __shared__ float red[256];
  long long idx = (long long)blockIdx.x * 256 + threadIdx.x;
  float am = 0.0f;
  if (idx < NTOT) {
    long long hidx;
    if (token) {
      int d = (int)(idx % Dd);
      long long t = idx / Dd;
      int s = (int)(t % Ss);
      int b = (int)(t / Ss);
      hidx = ((long long)b * Dd + d) * SP + s;
    } else hidx = idx;
    am = fabsf(h[hidx] + ident[idx]);
  }
  float bm = blockMax256(am, red);
  if (threadIdx.x == 0) atomicMaxPosF(slot, bm);
}

__global__ __launch_bounds__(256) void k_resid_requant(
    const float* __restrict__ h, const float* __restrict__ ident,
    const float* __restrict__ idsfPtr, int id_is_scale,
    const float* __restrict__ aAbs, const float* __restrict__ wsf,
    const float* __restrict__ sAbs, float* __restrict__ out,
    int token, float* __restrict__ sfout)
{
  long long idx = (long long)blockIdx.x * 256 + threadIdx.x;
  if (idx >= NTOT) return;
  float sf   = sf_from_abs(sAbs[0]);
  float idsf = id_is_scale ? idsfPtr[0] : sf_from_abs(idsfPtr[0]);
  float asf  = sf_from_abs(aAbs[0]);
  int d = (int)(idx % Dd);
  long long t = idx / Dd;
  int s = (int)(t % Ss);
  int b = (int)(t / Ss);
  long long hidx; int ncol;
  if (token) { hidx = ((long long)b * Dd + d) * SP + s; ncol = s; }
  else       { hidx = idx;                              ncol = d; }
  float pre = asf * wsf[ncol];
  float z = rintf(h[hidx] / pre) * (pre / sf)
          + rintf(ident[idx] / idsf) * (idsf / sf);
  out[idx] = clip8f(rintf(z)) * sf;
  if (sfout && idx == 0) sfout[0] = sf;
}

extern "C" void kernel_launch(void* const* d_in, const int* in_sizes, int n_in,
                              void* d_out, int out_size, void* d_ws, size_t ws_size,
                              hipStream_t stream) {
  (void)in_sizes; (void)n_in; (void)out_size; (void)ws_size;

  const float* x     = (const float*)d_in[0];
  const float* sf_in = (const float*)d_in[1];
  const float* g1    = (const float*)d_in[2];
  const float* bn1   = (const float*)d_in[3];
  const float* g2    = (const float*)d_in[4];
  const float* bn2   = (const float*)d_in[5];
  const float* Wt1   = (const float*)d_in[6];
  const float* bt1   = (const float*)d_in[7];
  const float* Wt2   = (const float*)d_in[8];
  const float* bt2   = (const float*)d_in[9];
  const float* Wc1   = (const float*)d_in[10];
  const float* bc1   = (const float*)d_in[11];
  const float* Wc2   = (const float*)d_in[12];
  const float* bc2   = (const float*)d_in[13];
  float* out = (float*)d_out;

  auto al = [](size_t v) { return (v + 255) & ~(size_t)255; };
  char* base = (char*)d_ws;
  size_t o = 0;
  float*       scal  = (float*)(base + o);       o += al(64 * 4);
  float*       lnbuf = (float*)(base + o);       o += al((size_t)NTOT * 4);
  float*       x1buf = (float*)(base + o);       o += al((size_t)NTOT * 4);
  signed char* qaA   = (signed char*)(base + o); o += al((size_t)BD * SP);
  signed char* qaB   = (signed char*)(base + o); o += al((size_t)BS * 3072);
  signed char* qw    = (signed char*)(base + o); o += al((size_t)3072 * 768);
  float*       wsf   = (float*)(base + o);       o += al(3072 * 4);
  float*       hA    = (float*)(base + o);       o += al((size_t)BS * 3072 * 4);
  float*       hB    = (float*)(base + o);       o += al((size_t)BD * SP * 4);

  const long long nTok1 = (long long)BD * 384;
  const long long nChan = (long long)BS * 3072;

  k_zero_scal<<<1, 64, 0, stream>>>(scal);

  k_ln_absmax<<<BS, 256, 0, stream>>>(x, g1, bn1, lnbuf, scal + 0);
  k_quantA_token<<<(unsigned)((long long)BD * SP / 256), 256, 0, stream>>>(lnbuf, scal + 0, qaA);

  k_wquant<<<384, 256, 0, stream>>>(Wt1, Ss, SP, 384, 384, qw, wsf);
  k_gemm_i8<<<dim3(3, 768), 256, 0, stream>>>(qaA, qw, bt1, wsf, scal + 0, hA, scal + 1, SP, 384);

  k_requant_gelu<<<(unsigned)(nTok1 / 256), 256, 0, stream>>>(hA, scal + 1, scal + 2, nTok1);
  k_quantA_flat<<<(unsigned)(nTok1 / 256), 256, 0, stream>>>(hA, scal + 2, qaB, 384 / 64, nTok1);

  k_wquant<<<SP, 256, 0, stream>>>(Wt2, 384, 384, SP, Ss, qw, wsf);
  k_gemm_i8<<<dim3(2, 768), 256, 0, stream>>>(qaB, qw, bt2, wsf, scal + 2, hB, scal + 3, 384, SP);

  k_resid_absmax<<<(unsigned)(NTOT / 256), 256, 0, stream>>>(hB, x, scal + 4, 1);
  k_resid_requant<<<(unsigned)(NTOT / 256), 256, 0, stream>>>(
      hB, x, sf_in, 1, scal + 2, wsf, scal + 4, x1buf, 1, nullptr);

  k_ln_absmax<<<BS, 256, 0, stream>>>(x1buf, g2, bn2, lnbuf, scal + 5);
  k_quantA_flat<<<(unsigned)(NTOT / 256), 256, 0, stream>>>(lnbuf, scal + 5, qaA, Dd / 64, NTOT);

  k_wquant<<<3072, 256, 0, stream>>>(Wc1, Dd, Dd, 3072, 3072, qw, wsf);
  k_gemm_i8<<<dim3(24, 196), 256, 0, stream>>>(qaA, qw, bc1, wsf, scal + 5, hA, scal + 6, Dd, 3072);

  k_requant_gelu<<<(unsigned)(nChan / 256), 256, 0, stream>>>(hA, scal + 6, scal + 7, nChan);
  k_quantA_flat<<<(unsigned)(nChan / 256), 256, 0, stream>>>(hA, scal + 7, qaB, 3072 / 64, nChan);

  k_wquant<<<768, 256, 0, stream>>>(Wc2, 3072, 3072, Dd, Dd, qw, wsf);
  k_gemm_i8<<<dim3(6, 196), 256, 0, stream>>>(qaB, qw, bc2, wsf, scal + 7, hB, scal + 8, 3072, Dd);

  k_resid_absmax<<<(unsigned)(NTOT / 256), 256, 0, stream>>>(hB, x1buf, scal + 9, 0);
  k_resid_requant<<<(unsigned)(NTOT / 256), 256, 0, stream>>>(
      hB, x1buf, scal + 4, 0, scal + 7, wsf, scal + 9, out, 0, out + NTOT);
}
